// WaveformDecoder_90933047591256
// MI455X (gfx1250) — matrix operand order repaired and checked by reading, not compiled, not run on hardware
//
#include <hip/hip_runtime.h>
#include <hip/hip_bf16.h>

typedef __attribute__((ext_vector_type(16))) __bf16 v16bf;
typedef __attribute__((ext_vector_type(8)))  float  v8f;

#define B_   2048
#define LAT_ 32
#define H_   256
#define D_   20
#define T_   512

// ---- swizzled weight workspace layout (bf16 elements) ----
// Fragment (nt, kt) of a K x N B-operand is 512 contiguous bf16:
//   lane L(0..31) -> n = nt*16 + L%16 ; 16 contiguous values s -> k = kt*32 + 8*(L/16) + (s&7) + 16*(s>>3)
#define W0_NKT 9           // layer0: K = 32 (x, padded from 20) + 256 (h0)
#define W1_NKT 16          // layer1: K = 256 (h0) + 256 (h1)
#define OFF_W0 0
#define ELEMS_W0 (64*W0_NKT*512)
#define OFF_W1 (OFF_W0 + ELEMS_W0)
#define ELEMS_W1 (64*W1_NKT*512)
#define OFF_P  (OFF_W1 + ELEMS_W1)
#define ELEMS_P (2*8*512)
#define OFF_LH (OFF_P + ELEMS_P)
#define ELEMS_LH (32*512)
#define OFF_LC (OFF_LH + ELEMS_LH)
#define ELEMS_LC (32*512)
#define TOT_ELEMS (OFF_LC + ELEMS_LC)

#define XSTR 40            // 32 + 8 pad (bank spread), 16B-aligned rows
#define HSTR 264           // 256 + 8 pad

__device__ __forceinline__ v8f zerov(){
  v8f z = {0.f,0.f,0.f,0.f,0.f,0.f,0.f,0.f}; return z;
}

// B fragment: 32 contiguous bytes per lane (pre-swizzled in workspace)
__device__ __forceinline__ v16bf ldb32(const __bf16* p){
  float4 lo = *(const float4*)(p);
  float4 hi = *(const float4*)(p + 8);
  v16bf r;
  __builtin_memcpy(&r, &lo, 16);
  __builtin_memcpy(((char*)&r) + 16, &hi, 16);
  return r;
}

// A fragment from row-major LDS: elems [k0..k0+7] and [k0+16..k0+23]
__device__ __forceinline__ v16bf lda32(const __bf16* p){
  float4 lo = *(const float4*)(p);
  float4 hi = *(const float4*)(p + 16);
  v16bf r;
  __builtin_memcpy(&r, &lo, 16);
  __builtin_memcpy(((char*)&r) + 16, &hi, 16);
  return r;
}

__device__ __forceinline__ v8f wmma_bf16(v16bf a, v16bf b, v8f c){
  return __builtin_amdgcn_wmma_f32_16x16x32_bf16(false, a, false, b,
                                                 (short)0, c, false, false);
}

__device__ __forceinline__ float sigm(float x){
  return 1.0f / (1.0f + __expf(-x));
}
__device__ __forceinline__ float tanh_(float x){
  x = fminf(fmaxf(x, -15.0f), 15.0f);
  float e = __expf(2.0f * x);
  return (e - 1.0f) / (e + 1.0f);
}

// LSTM elementwise update. acc[j]: j/2 selects gate (i,f,g,o), j&1 selects the
// 16-col subtile. Wave w owns columns [32w, 32w+32). c stays in registers.
__device__ __forceinline__ void lstm_update(const v8f* acc, v8f* cr,
    const float* bias, __bf16* h_lds, int wave, int lcol, int khalf)
{
#pragma unroll
  for (int jj = 0; jj < 2; ++jj){
    const int ncol = 32*wave + 16*jj + lcol;
    const float bi = bias[      ncol];
    const float bf = bias[256 + ncol];
    const float bg = bias[512 + ncol];
    const float bo = bias[768 + ncol];
#pragma unroll
    for (int v = 0; v < 8; ++v){
      const float ig = sigm (acc[    jj][v] + bi);
      const float fg = sigm (acc[2 + jj][v] + bf);
      const float gg = tanh_(acc[4 + jj][v] + bg);
      const float og = sigm (acc[6 + jj][v] + bo);
      const float c  = fg * cr[jj][v] + ig * gg;
      cr[jj][v] = c;
      h_lds[(v + 8*khalf)*HSTR + ncol] = (__bf16)(og * tanh_(c));
    }
  }
}

// ---------- one-time weight swizzle into WMMA B-operand layout ----------
__global__ __launch_bounds__(256)
void wd_prep(const float* __restrict__ w_lh, const float* __restrict__ w_lc,
             const float* __restrict__ w_ih0, const float* __restrict__ w_hh0,
             const float* __restrict__ w_ih1, const float* __restrict__ w_hh1,
             const float* __restrict__ w_proj, __bf16* __restrict__ ws)
{
  const int idx = blockIdx.x * 256 + threadIdx.x;
  if (idx >= TOT_ELEMS) return;
  const int r    = idx & 511;
  const int lane = ((r >> 5) & 8) + (r & 7) + ((r & 8) << 1);
  const int j    = (r >> 4) & 15;
  float v;
  if (idx < OFF_W1){                       // layer0: [x(pad 32) | h0] -> 4H
    const int frag = idx >> 9;
    const int nt = frag / W0_NKT, kt = frag - nt * W0_NKT;
    const int k = kt*32 + lane, n = nt*16 + j;
    v = (k < 32) ? ((k < D_) ? w_ih0[n*D_ + k] : 0.0f)
                 : w_hh0[n*H_ + (k - 32)];
  } else if (idx < OFF_P){                 // layer1: [h0 | h1] -> 4H
    const int frag = (idx - OFF_W1) >> 9;
    const int nt = frag >> 4, kt = frag & 15;
    const int k = kt*32 + lane, n = nt*16 + j;
    v = (k < H_) ? w_ih1[n*H_ + k] : w_hh1[n*H_ + (k - H_)];
  } else if (idx < OFF_LH){                // projection (N padded 20->32)
    const int frag = (idx - OFF_P) >> 9;
    const int nt = frag >> 3, kt = frag & 7;
    const int k = kt*32 + lane, n = nt*16 + j;
    v = (n < D_) ? w_proj[n*H_ + k] : 0.0f;
  } else if (idx < OFF_LC){                // w_lh (K = LAT = 32)
    const int frag = (idx - OFF_LH) >> 9;
    const int n = frag*16 + j;
    v = w_lh[n*LAT_ + lane];
  } else {                                 // w_lc
    const int frag = (idx - OFF_LC) >> 9;
    const int n = frag*16 + j;
    v = w_lc[n*LAT_ + lane];
  }
  ws[idx] = (__bf16)v;
}

// ---------- persistent per-batch-tile decoder ----------
__global__ __launch_bounds__(256)
void wd_decoder(const float* __restrict__ z,
                const float* __restrict__ b_lh, const float* __restrict__ b_lc,
                const float* __restrict__ b_ih0, const float* __restrict__ b_hh0,
                const float* __restrict__ b_ih1, const float* __restrict__ b_hh1,
                const float* __restrict__ b_proj,
                const __bf16* __restrict__ ws, float* __restrict__ out)
{
  __shared__ __align__(16) __bf16 x_lds[16*XSTR];
  __shared__ __align__(16) __bf16 h0_lds[16*HSTR];
  __shared__ __align__(16) __bf16 h1_lds[16*HSTR];
  __shared__ float bias0_lds[1024];
  __shared__ float bias1_lds[1024];
  __shared__ float bproj_lds[32];

  const int tid   = threadIdx.x;
  const int lane  = tid & 31;          // wave32
  const int wave  = tid >> 5;          // 8 waves
  const int khalf = lane >> 4;
  const int lcol  = lane & 15;
  const int b0    = blockIdx.x * 16;   // this WG's batch rows

  const __bf16* W0  = ws + OFF_W0;
  const __bf16* W1  = ws + OFF_W1;
  const __bf16* WP  = ws + OFF_P;
  const __bf16* WLH = ws + OFF_LH;
  const __bf16* WLC = ws + OFF_LC;

  for (int i = tid; i < 1024; i += 256){
    bias0_lds[i] = b_ih0[i] + b_hh0[i];
    bias1_lds[i] = b_ih1[i] + b_hh1[i];
  }
  if (tid < 32) bproj_lds[tid] = (tid < D_) ? b_proj[tid] : 0.0f;
  for (int i = tid; i < 16*XSTR; i += 256) x_lds[i] = (__bf16)0.0f;  // x0 = 0
  __syncthreads();

  v8f c0r[2], c1r[2];

  // ---- init: h/c = z @ w_l{h,c}.T + b (one 16x16x32 WMMA per tile) ----
  {
    const float* zrow = z + (size_t)(b0 + lcol) * LAT_;
    v16bf az;
#pragma unroll
    for (int j = 0; j < 8; ++j){
      az[j]     = (__bf16)zrow[8*khalf + j];
      az[8 + j] = (__bf16)zrow[16 + 8*khalf + j];
    }
#pragma unroll
    for (int q = 0; q < 4; ++q){
      const int nt = (q >> 1)*16 + 2*wave + (q & 1);
      v8f hv = wmma_bf16(az, ldb32(WLH + nt*512 + lane*16), zerov());
      v8f cv = wmma_bf16(az, ldb32(WLC + nt*512 + lane*16), zerov());
      const int ng    = nt*16 + lcol;
      const float bh  = b_lh[ng];
      const float bc  = b_lc[ng];
      const int layer = q >> 1;
      const int ncol  = ng - 256*layer;
      __bf16* hl = layer ? h1_lds : h0_lds;
      v8f*    cr = layer ? c1r    : c0r;
#pragma unroll
      for (int v = 0; v < 8; ++v){
        hl[(v + 8*khalf)*HSTR + ncol] = (__bf16)(hv[v] + bh);
        cr[q & 1][v] = cv[v] + bc;
      }
    }
  }
  __syncthreads();

  // ---- 512 sequential timesteps, batch tile fully private to this WG ----
#pragma unroll 1
  for (int t = 0; t < T_; ++t){
    v8f acc[8];

    // layer 0: gates = [x | h0] @ W0^T  (K = 32 + 256)
#pragma unroll
    for (int j = 0; j < 8; ++j) acc[j] = zerov();
    {
      v16bf a = lda32(x_lds + lcol*XSTR + 8*khalf);
#pragma unroll
      for (int j = 0; j < 8; ++j){
        const int nt = (j >> 1)*16 + 2*wave + (j & 1);
        acc[j] = wmma_bf16(a, ldb32(W0 + (nt*W0_NKT + 0)*512 + lane*16), acc[j]);
      }
    }
#pragma unroll
    for (int kt = 1; kt < W0_NKT; ++kt){
      v16bf a = lda32(h0_lds + lcol*HSTR + (kt - 1)*32 + 8*khalf);
#pragma unroll
      for (int j = 0; j < 8; ++j){
        const int nt = (j >> 1)*16 + 2*wave + (j & 1);
        acc[j] = wmma_bf16(a, ldb32(W0 + (nt*W0_NKT + kt)*512 + lane*16), acc[j]);
      }
    }
    __syncthreads();                       // all h0 reads done
    lstm_update(acc, c0r, bias0_lds, h0_lds, wave, lcol, khalf);
    __syncthreads();                       // new h0 visible

    // layer 1: gates = [h0 | h1] @ W1^T  (K = 512)
#pragma unroll
    for (int j = 0; j < 8; ++j) acc[j] = zerov();
#pragma unroll
    for (int kt = 0; kt < W1_NKT; ++kt){
      const __bf16* asrc = (kt < 8)
          ? (h0_lds + lcol*HSTR + kt*32 + 8*khalf)
          : (h1_lds + lcol*HSTR + (kt - 8)*32 + 8*khalf);
      v16bf a = lda32(asrc);
      if (kt + 1 < W1_NKT)                 // stream next K-tile (L2-resident)
        __builtin_prefetch(W1 + (((2*wave)*W1_NKT) + kt + 1)*512 + lane*16, 0, 0);
#pragma unroll
      for (int j = 0; j < 8; ++j){
        const int nt = (j >> 1)*16 + 2*wave + (j & 1);
        acc[j] = wmma_bf16(a, ldb32(W1 + (nt*W1_NKT + kt)*512 + lane*16), acc[j]);
      }
    }
    __syncthreads();                       // all h1 reads done
    lstm_update(acc, c1r, bias1_lds, h1_lds, wave, lcol, khalf);
    __syncthreads();                       // new h1 visible

    // projection: y = h1 @ w_proj^T + b  (waves 0,1 only; N = 20)
    if (wave < 2){
      v8f py = zerov();
#pragma unroll
      for (int kt = 0; kt < 8; ++kt){
        v16bf a = lda32(h1_lds + lcol*HSTR + kt*32 + 8*khalf);
        py = wmma_bf16(a, ldb32(WP + (wave*8 + kt)*512 + lane*16), py);
      }
      const int n = wave*16 + lcol;
      if (n < D_){
        const float bp = bproj_lds[n];
#pragma unroll
        for (int v = 0; v < 8; ++v){
          const float yv = py[v] + bp;
          const int m = v + 8*khalf;
          out[(size_t)(b0 + m)*(T_*D_) + (size_t)t*D_ + n] = yv;
          x_lds[m*XSTR + n] = (__bf16)yv;  // y feeds next step's x
        }
      }
    }
    __syncthreads();                       // x ready for t+1
  }
}

extern "C" void kernel_launch(void* const* d_in, const int* in_sizes, int n_in,
                              void* d_out, int out_size, void* d_ws, size_t ws_size,
                              hipStream_t stream) {
  const float* z      = (const float*)d_in[0];
  // d_in[1] target_lengths, d_in[2] max_len: unused (reference emits full T)
  const float* w_lh   = (const float*)d_in[3];
  const float* b_lh   = (const float*)d_in[4];
  const float* w_lc   = (const float*)d_in[5];
  const float* b_lc   = (const float*)d_in[6];
  const float* w_ih0  = (const float*)d_in[7];
  const float* w_hh0  = (const float*)d_in[8];
  const float* b_ih0  = (const float*)d_in[9];
  const float* b_hh0  = (const float*)d_in[10];
  const float* w_ih1  = (const float*)d_in[11];
  const float* w_hh1  = (const float*)d_in[12];
  const float* b_ih1  = (const float*)d_in[13];
  const float* b_hh1  = (const float*)d_in[14];
  const float* w_proj = (const float*)d_in[15];
  const float* b_proj = (const float*)d_in[16];
  __bf16* ws = (__bf16*)d_ws;
  float* out = (float*)d_out;

  wd_prep<<<(TOT_ELEMS + 255)/256, 256, 0, stream>>>(
      w_lh, w_lc, w_ih0, w_hh0, w_ih1, w_hh1, w_proj, ws);
  wd_decoder<<<B_/16, 256, 0, stream>>>(
      z, b_lh, b_lc, b_ih0, b_hh0, b_ih1, b_hh1, b_proj, ws, out);
}
